// SCA2LSTM_54975581389304
// MI455X (gfx1250) — matrix operand order repaired and checked by reading, not compiled, not run on hardware
//
#include <hip/hip_runtime.h>

// ---------------------------------------------------------------------------
// SCA2LSTM fused pipeline for gfx1250 (MI455X), wave32 + WMMA bf16.
// B=256, T=512, NF=32, D1=32, H=256, E=32.
// Out = [pred(256) | w(256*512*32)] fp32.
// ---------------------------------------------------------------------------

typedef __bf16 bf16;
typedef __attribute__((ext_vector_type(16))) __bf16 v16bf;
typedef __attribute__((ext_vector_type(8)))  float  v8f;

#define WMMA_BF16(a, b, c) \
  __builtin_amdgcn_wmma_f32_16x16x32_bf16(false, (a), false, (b), (short)0, (c), false, false)

#define B_  256
#define T_  512
#define NF_ 32
#define H_  256

__device__ __forceinline__ float clipf_(float x, float lo, float hi) {
  return fminf(fmaxf(x, lo), hi);
}
// Fast reciprocal: single v_rcp_f32.
__device__ __forceinline__ float frcp_(float x)  { return __builtin_amdgcn_rcpf(x); }

// Native CDNA5 v_tanh_f32 when the builtin exists; exp/rcp fallback otherwise.
#if __has_builtin(__builtin_amdgcn_tanhf)
__device__ __forceinline__ float tanh_(float x)  { return __builtin_amdgcn_tanhf(x); }
__device__ __forceinline__ float sig_(float x)   {
  return __builtin_fmaf(0.5f, __builtin_amdgcn_tanhf(0.5f * x), 0.5f);
}
#else
__device__ __forceinline__ float tanh_(float x)  { return 2.0f * frcp_(1.0f + __expf(-2.0f * x)) - 1.0f; }
__device__ __forceinline__ float sig_(float x)   { return frcp_(1.0f + __expf(-x)); }
#endif
__device__ __forceinline__ float lrelu_(float x) { return x > 0.0f ? x : 0.01f * x; }

// Per-wave LDS producer->consumer ordering (DS ops are in-order per wave;
// the wait + scheduling barrier keeps the compiler from reordering).
__device__ __forceinline__ void wave_lds_sync() {
  __builtin_amdgcn_wave_barrier();
  asm volatile("s_wait_dscnt 0" ::: "memory");
  __builtin_amdgcn_wave_barrier();
}

// Dynamic LDS layout (bytes):
//   [0      .. 65536)  W1 bf16   (1024 x 32)
//   [65536  .. 81920)  Wg1 bf16  (32 x 256)
//   [81920  .. 83968)  Wg2 bf16  (32 x 32)
//   [83968  .. 88064)  b1  f32   (1024)
//   [88064  .. 88192)  bg1 f32   (32)
//   [88192  .. 88320)  bg2 f32   (32)
//   [88320  .. +8*11264) per-wave scratch: h1 bf16(16x256)=8192B,
//                         hid bf16(16x32)=1024B, logits f32(16x32)=2048B
#define NWAVE_MAIN 8
#define SMEM_BYTES (88320 + NWAVE_MAIN * 11264)

extern "C" __global__ void __launch_bounds__(256)
k_sca2_main(const float* __restrict__ x1g,    // lstm1_input (B,T,32)
            const float* __restrict__ miss,   // missing_bool (B,T,32)
            const float* __restrict__ W1,     // (1024,32)
            const float* __restrict__ b1,     // (1024)
            const float* __restrict__ Wg1,    // (32,256)
            const float* __restrict__ bg1,    // (32)
            const float* __restrict__ Wg2,    // (32,32)
            const float* __restrict__ bg2,    // (32)
            float* __restrict__ wout)         // (B,T,32)
{
  extern __shared__ char smem[];
  bf16*  W1s  = (bf16*)smem;                       // 32768 elems
  bf16*  Wg1s = W1s + 1024 * 32;                   // 8192 elems
  bf16*  Wg2s = Wg1s + 32 * 256;                   // 1024 elems
  float* b1s  = (float*)(smem + 83968);            // 1024
  float* bg1s = b1s + 1024;                        // 32
  float* bg2s = bg1s + 32;                         // 32
  char*  wsbase = smem + 88320;

  const int tid = threadIdx.x;
  // Cooperative weight staging (fp32 -> bf16)
  for (int i = tid; i < 1024 * 32; i += blockDim.x) W1s[i]  = (bf16)W1[i];
  for (int i = tid; i < 32 * 256;  i += blockDim.x) Wg1s[i] = (bf16)Wg1[i];
  for (int i = tid; i < 32 * 32;   i += blockDim.x) Wg2s[i] = (bf16)Wg2[i];
  for (int i = tid; i < 1024;      i += blockDim.x) b1s[i]  = b1[i];
  if (tid < 32) { bg1s[tid] = bg1[tid]; bg2s[tid] = bg2[tid]; }
  __syncthreads();

  const int wave = tid >> 5;
  const int lane = tid & 31;
  const int lrow = lane & 15;   // data row within tile (A/C layouts)
  const int lhi  = lane >> 4;   // lane-half selector for K/M split

  bf16*  h1s  = (bf16*)(wsbase + wave * 11264);
  bf16*  hids = h1s + 16 * 256;
  float* lg   = (float*)(hids + 16 * 32);

  const int nWaves = (gridDim.x * blockDim.x) >> 5;
  const int gw = (blockIdx.x * blockDim.x + tid) >> 5;

  for (int tile = gw; tile < (B_ * T_) / 16; tile += nWaves) {
    const int r0 = tile * 16;

    // ---- A fragment: x1 tile (16 x 32), clipped to +-3, bf16 --------------
    v16bf a1;
    {
      const float* xr = x1g + (size_t)(r0 + lrow) * 32 + lhi * 8;
      #pragma unroll
      for (int e = 0; e < 8; ++e) {
        a1[e]     = (bf16)clipf_(xr[e],      -3.0f, 3.0f);
        a1[8 + e] = (bf16)clipf_(xr[16 + e], -3.0f, 3.0f);
      }
    }

    // ---- GEMM1 + gates: h1 = clip(sig(o)*tanh(sig(i)*tanh(c)), +-5) ------
    #pragma unroll 1
    for (int j = 0; j < 16; ++j) {
      const int n = j * 16 + lrow;               // h-index column of this lane
      const bf16* pi = W1s + (size_t)(0   + n) * 32 + lhi * 8;
      const bf16* pc = W1s + (size_t)(512 + n) * 32 + lhi * 8;
      const bf16* po = W1s + (size_t)(768 + n) * 32 + lhi * 8;
      v16bf bi, bc, bo;
      #pragma unroll
      for (int e = 0; e < 16; ++e) { const int kk = e < 8 ? e : 8 + e; bi[e] = pi[kk]; bc[e] = pc[kk]; bo[e] = po[kk]; }
      v8f ci = {}, cc = {}, co = {};
      ci = WMMA_BF16(a1, bi, ci);
      cc = WMMA_BF16(a1, bc, cc);
      co = WMMA_BF16(a1, bo, co);
      #pragma unroll
      for (int r = 0; r < 8; ++r) {
        const int m = r + lhi * 8;
        float gi = ci[r] + b1s[n];
        float gc = cc[r] + b1s[512 + n];
        float go = co[r] + b1s[768 + n];
        float h  = sig_(go) * tanh_(sig_(gi) * tanh_(gc));
        h1s[m * 256 + n] = (bf16)clipf_(h, -5.0f, 5.0f);
      }
    }
    wave_lds_sync();

    // ---- GEMM2: hid = leaky(h1 @ Wg1^T + bg1), 16x32 ----------------------
    v8f c20 = {}, c21 = {};
    #pragma unroll 1
    for (int ks = 0; ks < 8; ++ks) {
      v16bf a2;
      const bf16* hp = h1s + lrow * 256 + ks * 32 + lhi * 8;
      #pragma unroll
      for (int e = 0; e < 8; ++e) { a2[e] = hp[e]; a2[8 + e] = hp[16 + e]; }
      const bf16* q0 = Wg1s + (size_t)(lrow)      * 256 + ks * 32 + lhi * 8;
      const bf16* q1 = Wg1s + (size_t)(16 + lrow) * 256 + ks * 32 + lhi * 8;
      v16bf b20, b21;
      #pragma unroll
      for (int e = 0; e < 16; ++e) { const int kk = e < 8 ? e : 8 + e; b20[e] = q0[kk]; b21[e] = q1[kk]; }
      c20 = WMMA_BF16(a2, b20, c20);
      c21 = WMMA_BF16(a2, b21, c21);
    }
    #pragma unroll
    for (int r = 0; r < 8; ++r) {
      const int m = r + lhi * 8;
      hids[m * 32 + lrow]      = (bf16)lrelu_(c20[r] + bg1s[lrow]);
      hids[m * 32 + 16 + lrow] = (bf16)lrelu_(c21[r] + bg1s[16 + lrow]);
    }
    wave_lds_sync();

    // ---- GEMM3: logits = hid @ Wg2^T + bg2, 16x32 -------------------------
    {
      v16bf a3;
      const bf16* hp = hids + lrow * 32 + lhi * 8;
      #pragma unroll
      for (int e = 0; e < 8; ++e) { a3[e] = hp[e]; a3[8 + e] = hp[16 + e]; }
      const bf16* q0 = Wg2s + (size_t)(lrow)      * 32 + lhi * 8;
      const bf16* q1 = Wg2s + (size_t)(16 + lrow) * 32 + lhi * 8;
      v16bf b30, b31;
      #pragma unroll
      for (int e = 0; e < 16; ++e) { const int kk = e < 8 ? e : 8 + e; b30[e] = q0[kk]; b31[e] = q1[kk]; }
      v8f c30 = {}, c31 = {};
      c30 = WMMA_BF16(a3, b30, c30);
      c31 = WMMA_BF16(a3, b31, c31);
      #pragma unroll
      for (int r = 0; r < 8; ++r) {
        const int m = r + lhi * 8;
        lg[m * 32 + lrow]      = c30[r] + bg2s[lrow];
        lg[m * 32 + 16 + lrow] = c31[r] + bg2s[16 + lrow];
      }
    }
    wave_lds_sync();

    // ---- softmax + mask + renorm + clip -> w ------------------------------
    if (lane < 16) {
      const int gr = r0 + lane;
      float v[32];
      float mx = -1e30f;
      #pragma unroll
      for (int f = 0; f < 32; ++f) { v[f] = lg[lane * 32 + f]; mx = fmaxf(mx, v[f]); }
      float s = 0.0f;
      #pragma unroll
      for (int f = 0; f < 32; ++f) { v[f] = __expf(v[f] - mx); s += v[f]; }
      const float inv = frcp_(s);
      const float* mb = miss + (size_t)gr * 32;
      float s2 = 0.0f;
      #pragma unroll
      for (int f = 0; f < 32; ++f) { v[f] = v[f] * inv * mb[f]; s2 += v[f]; }
      const float inv2 = frcp_(s2 + 1e-8f);
      float* wo = wout + (size_t)gr * 32;
      #pragma unroll
      for (int f = 0; f < 32; ++f) wo[f] = clipf_(v[f] * inv2, 1e-6f, 0.5f);
    }
    wave_lds_sync();
  }
}

// ---------------------------------------------------------------------------
// Last-timestep path: h2 gates for t=T-1 only (256 rows), then the MLP head.
// Reads w[:,T-1,:] back from d_out (stream-ordered after k_sca2_main).
// ---------------------------------------------------------------------------
extern "C" __global__ void __launch_bounds__(128)
k_sca2_last(const float* __restrict__ seqf,   // (B,T,32)
            const int*   __restrict__ bid,    // (B,)
            const float* __restrict__ bemb,   // (NB,32)
            const float* __restrict__ W2,     // (1024,64)
            const float* __restrict__ b2,     // (1024)
            const float* __restrict__ Wp1,    // (16,256)
            const float* __restrict__ bp1,    // (16)
            const float* __restrict__ Wp2,    // (1,16)
            const float* __restrict__ bp2,    // (1)
            const float* __restrict__ wfull,  // (B,T,32) = d_out + 256
            float* __restrict__ pred)         // (B,)
{
  __shared__ bf16 h2all[4][16 * 256];

  const int tid  = threadIdx.x;
  const int wave = tid >> 5;
  const int lane = tid & 31;
  const int lrow = lane & 15;
  const int lhi  = lane >> 4;

  bf16* h2s = h2all[wave];
  const int tile = blockIdx.x * 4 + wave;        // 0..15 with grid=4
  const int b = tile * 16 + lrow;

  // ---- A fragments: x2 = [wf | emb], two K=32 slices ----------------------
  v16bf aWF, aEM;
  {
    const size_t base = ((size_t)b * T_ + (T_ - 1)) * 32;
    const float* sr = seqf  + base;
    const float* wr = wfull + base;
    const float* er = bemb + (size_t)bid[b] * 32;
    const int kb = lhi * 8;
    #pragma unroll
    for (int e = 0; e < 8; ++e) {
      int f0 = kb + e, f1 = kb + 16 + e;
      aWF[e]     = (bf16)clipf_(clipf_(sr[f0], -3.0f, 3.0f) * wr[f0], -3.0f, 3.0f);
      aWF[8 + e] = (bf16)clipf_(clipf_(sr[f1], -3.0f, 3.0f) * wr[f1], -3.0f, 3.0f);
      aEM[e]     = (bf16)clipf_(er[f0], -3.0f, 3.0f);
      aEM[8 + e] = (bf16)clipf_(er[f1], -3.0f, 3.0f);
    }
  }

  // ---- GEMM4 gates (i,c,o), K=64 as two K=32 WMMAs ------------------------
  #pragma unroll 1
  for (int j = 0; j < 16; ++j) {
    const int n = j * 16 + lrow;
    v8f ci = {}, cc = {}, co = {};
    #pragma unroll
    for (int kt = 0; kt < 2; ++kt) {
      const float* pi = W2 + (size_t)(0   + n) * 64 + kt * 32 + lhi * 8;
      const float* pc = W2 + (size_t)(512 + n) * 64 + kt * 32 + lhi * 8;
      const float* po = W2 + (size_t)(768 + n) * 64 + kt * 32 + lhi * 8;
      v16bf bi, bc, bo;
      #pragma unroll
      for (int e = 0; e < 16; ++e) {
        const int kk = e < 8 ? e : 8 + e;
        bi[e] = (bf16)pi[kk]; bc[e] = (bf16)pc[kk]; bo[e] = (bf16)po[kk];
      }
      const v16bf A = kt ? aEM : aWF;
      ci = WMMA_BF16(A, bi, ci);
      cc = WMMA_BF16(A, bc, cc);
      co = WMMA_BF16(A, bo, co);
    }
    #pragma unroll
    for (int r = 0; r < 8; ++r) {
      const int m = r + lhi * 8;
      float gi = ci[r] + b2[n];
      float gc = cc[r] + b2[512 + n];
      float go = co[r] + b2[768 + n];
      float h  = sig_(go) * tanh_(sig_(gi) * tanh_(gc));
      h2s[m * 256 + n] = (bf16)clipf_(h, -5.0f, 5.0f);
    }
  }
  wave_lds_sync();

  // ---- head: pred = clip(leaky(h2 @ Wp1^T + bp1) @ Wp2^T + bp2, 0, 100) ---
  if (lane < 16) {
    const bf16* hr = h2s + lane * 256;
    float p = 0.0f;
    #pragma unroll 1
    for (int k = 0; k < 16; ++k) {
      float s = bp1[k];
      const float* wp = Wp1 + k * 256;
      #pragma unroll 8
      for (int h = 0; h < 256; ++h) s += (float)hr[h] * wp[h];
      p += lrelu_(s) * Wp2[k];
    }
    pred[tile * 16 + lane] = clipf_(p + bp2[0], 0.0f, 100.0f);
  }
}

// ---------------------------------------------------------------------------
extern "C" void kernel_launch(void* const* d_in, const int* in_sizes, int n_in,
                              void* d_out, int out_size, void* d_ws, size_t ws_size,
                              hipStream_t stream) {
  (void)in_sizes; (void)n_in; (void)out_size; (void)d_ws; (void)ws_size;
  const float* seqf = (const float*)d_in[0];
  const float* x1   = (const float*)d_in[1];
  const float* miss = (const float*)d_in[2];
  const int*   bid  = (const int*)  d_in[3];
  // d_in[4] = residual : dead code in the reference (LayerNorm result unused)
  const float* bemb = (const float*)d_in[5];
  const float* W1   = (const float*)d_in[6];
  const float* b1   = (const float*)d_in[7];
  const float* Wg1  = (const float*)d_in[8];
  const float* bg1  = (const float*)d_in[9];
  const float* Wg2  = (const float*)d_in[10];
  const float* bg2  = (const float*)d_in[11];
  const float* W2   = (const float*)d_in[12];
  const float* b2   = (const float*)d_in[13];
  const float* Wp1  = (const float*)d_in[14];
  const float* bp1  = (const float*)d_in[15];
  const float* Wp2  = (const float*)d_in[16];
  const float* bp2  = (const float*)d_in[17];
  // d_in[18]/d_in[19] = ln_g/ln_b : dead code

  float* pred = (float*)d_out;       // (B,1) = 256 floats
  float* wout = pred + B_;           // (B,T,32)

  k_sca2_main<<<128, 256, SMEM_BYTES, stream>>>(x1, miss, W1, b1, Wg1, bg1,
                                                Wg2, bg2, wout);
  k_sca2_last<<<4, 128, 0, stream>>>(seqf, bid, bemb, W2, b2, Wp1, bp1,
                                     Wp2, bp2, wout, pred);
}
